// TransMatch_86947317940362
// MI455X (gfx1250) — matrix operand order repaired and checked by reading, not compiled, not run on hardware
//
#include <hip/hip_runtime.h>
#include <hip/hip_bf16.h>

// MI455X / gfx1250, wave32. GEMMs via v_wmma_f32_16x16x32_bf16.
// K staged into LDS via the Tensor Data Mover (double-buffered) when available.

typedef __attribute__((ext_vector_type(16))) __bf16 v16bf;
typedef __attribute__((ext_vector_type(8)))  __bf16 v8bf;
typedef __attribute__((ext_vector_type(2)))  __bf16 v2bf;
typedef __attribute__((ext_vector_type(8)))  float  v8f;
typedef unsigned int u32x4 __attribute__((ext_vector_type(4)));
typedef int          i32x4 __attribute__((ext_vector_type(4)));
typedef int          i32x8 __attribute__((ext_vector_type(8)));

#define KL   8
#define KS   192
#define KD   512
#define KDFF 2048
#define KBS  16
#define KEPS 1e-5f

#if defined(__AMDGCN__) && __has_builtin(__builtin_amdgcn_tensor_load_to_lds)
#define USE_TDM 1
#else
#define USE_TDM 0
#endif

// ---------------------------------------------------------------- sigmoid(se)
__global__ void k_sigmoid(const float* __restrict__ se, float* __restrict__ sig, int n) {
    int i = blockIdx.x * blockDim.x + threadIdx.x;
    if (i < n) sig[i] = 1.0f / (1.0f + __expf(-se[i]));
}

// ------------------------------------------------- projection: Q/K = f @ W0^T
// grid: x=24 (3 m-blocks x 8 n-blocks), y=batch, z=which*8+l ; block=128 (4 waves)
// Q stored [l][b][s][d] (A-side row major); K stored transposed [l][b][d][s]
// (B-side row major) so the correlation kernel's LDS stage is a straight copy.
__global__ __launch_bounds__(128)
void k_project(const float* __restrict__ qfeat, const float* __restrict__ gfeat,
               const float* __restrict__ w0, const float* __restrict__ b0,
               __bf16* __restrict__ Qp, __bf16* __restrict__ KpT) {
    const int mb = blockIdx.x % 3, nb = blockIdx.x / 3;
    const int b  = blockIdx.y;
    const int l  = blockIdx.z & 7;
    const bool isK = blockIdx.z >= 8;
    const float* feat = isK ? gfeat : qfeat;

    __shared__ __attribute__((aligned(16))) __bf16 As[64][40];  // [m][k]
    __shared__ __attribute__((aligned(16))) __bf16 Bs[32][72];  // [k][n]

    const int tid = threadIdx.x;
    const int wid = tid >> 5, lane = tid & 31;
    const int khalf = lane >> 4, lrow = lane & 15;
    const int m0 = mb * 64, n0 = nb * 64;

    // A[m][k] = feat[b][l*512 + k][m0+m] (m contiguous in memory)
    const float* Abase = feat + ((size_t)b * (KD * KL) + (size_t)l * KD) * KS;
    const float* Bbase = w0 + (size_t)l * KD * KD;   // [n][k], k contiguous

    v8f acc[4] = {};

    for (int kk = 0; kk < KD; kk += 32) {
        { // stage A: thread -> pair of k-columns x 8 rows, packed b32 stores
            int kp = tid >> 3;             // 0..15 -> k = 2*kp, 2*kp+1
            int mstart = (tid & 7) * 8;    // 0..56
            const float* s0 = Abase + (size_t)(kk + 2 * kp) * KS + m0 + mstart;
            const float* s1 = s0 + KS;
            #pragma unroll
            for (int i = 0; i < 8; ++i) {
                v2bf p; p[0] = (__bf16)s0[i]; p[1] = (__bf16)s1[i];
                *(v2bf*)(&As[mstart + i][2 * kp]) = p;
            }
        }
        { // stage B transposed: thread -> pair of n-rows x 8 k, packed b32 stores
            int np = tid >> 2;             // 0..31 -> n = 2*np, 2*np+1
            int kstart = (tid & 3) * 8;    // 0..24
            const float* s0 = Bbase + (size_t)(n0 + 2 * np) * KD + kk + kstart;
            const float* s1 = s0 + KD;
            #pragma unroll
            for (int i = 0; i < 8; ++i) {
                v2bf p; p[0] = (__bf16)s0[i]; p[1] = (__bf16)s1[i];
                *(v2bf*)(&Bs[kstart + i][2 * np]) = p;
            }
        }
        __syncthreads();
        v16bf af;
        {
            const __bf16* ap = &As[16 * wid + lrow][0];
            v8bf lo = *(const v8bf*)(ap + 8 * khalf);
            v8bf hi = *(const v8bf*)(ap + 16 + 8 * khalf);
            #pragma unroll
            for (int i = 0; i < 8; ++i) { af[i] = lo[i]; af[8 + i] = hi[i]; }
        }
        #pragma unroll
        for (int j = 0; j < 4; ++j) {
            v16bf bfv;
            #pragma unroll
            for (int i = 0; i < 8; ++i) { bfv[i] = Bs[8 * khalf + i][16 * j + lrow]; bfv[8 + i] = Bs[16 + 8 * khalf + i][16 * j + lrow]; }
            acc[j] = __builtin_amdgcn_wmma_f32_16x16x32_bf16(
                false, af, false, bfv, (short)0, acc[j], false, false);
        }
        __syncthreads();
    }

    // C layout: vgpr r, lanes 0-15 -> m=r, lanes 16-31 -> m=8+r; n = lane&15
    if (!isK) {
        #pragma unroll
        for (int j = 0; j < 4; ++j) {
            int n = n0 + 16 * j + lrow;
            float bias = b0[l * KD + n];
            #pragma unroll
            for (int r = 0; r < 8; ++r) {
                int m = m0 + 16 * wid + 8 * khalf + r;
                Qp[(((size_t)l * KBS + b) * KS + m) * KD + n] = (__bf16)(acc[j][r] + bias);
            }
        }
    } else {
        // transposed: consecutive r -> consecutive s -> one packed 16B store
        #pragma unroll
        for (int j = 0; j < 4; ++j) {
            int n = n0 + 16 * j + lrow;              // d index
            float bias = b0[l * KD + n];
            v8bf vb;
            #pragma unroll
            for (int r = 0; r < 8; ++r) vb[r] = (__bf16)(acc[j][r] + bias);
            int m = m0 + 16 * wid + 8 * khalf;       // s start
            *(v8bf*)(KpT + (((size_t)l * KBS + b) * KD + n) * KS + m) = vb;
        }
    }
}

// -------------------- correlation 192x192 per (l,q,k) + mask + max reductions
// grid: x=256 (q*16+k), y=l ; block=384 (12 waves, each a 16-row t-strip)
// K chunk [32][192] copied contiguously global->LDS via TDM, double-buffered.
__global__ __launch_bounds__(384)
void k_corr(const __bf16* __restrict__ Qp, const __bf16* __restrict__ KpT,
            const float* __restrict__ sig, float* __restrict__ cat,
            float* __restrict__ bn1) {
    const int qk = blockIdx.x, l = blockIdx.y;
    const int q = qk >> 4, kb = qk & 15;
    const int tid = threadIdx.x;
    const int wid = tid >> 5, lane = tid & 31;
    const int khalf = lane >> 4, lrow = lane & 15;

    __shared__ __attribute__((aligned(16))) __bf16 KT[2][32][KS]; // [buf][d][s]
    __shared__ unsigned cmkey[KS];
    __shared__ float catv[2 * KS];
    __shared__ float red[24];

    const __bf16* Qb = Qp + (((size_t)l * KBS + q) * KS) * KD;
    const __bf16* Kg = KpT + (((size_t)l * KBS + kb) * KD) * KS;  // chunk kk at +kk*192

    v8f acc[12] = {};   // 12 tiles of 16x16 along s

    const unsigned ldsbase = (unsigned)(uintptr_t)(&KT[0][0][0]);
    (void)ldsbase;

    auto stage = [&](int kk, int buf) {
#if USE_TDM
        if (wid == 0) {
            unsigned long long ga = (unsigned long long)(uintptr_t)(Kg + (size_t)kk * KS);
            u32x4 g0;
            g0[0] = 1u;                                   // count=1, user descriptor
            g0[1] = ldsbase + (unsigned)buf * (32u * KS * 2u);  // lds_addr (bytes)
            g0[2] = (unsigned)(ga & 0xFFFFFFFFull);       // global_addr[31:0]
            g0[3] = (unsigned)((ga >> 32) & 0x1FFFFFFull) | (2u << 30); // [56:32] | type=2
            i32x8 g1;
            g1[0] = 1 << 16;                 // data_size = 2 bytes
            g1[1] = (int)(6144u << 16);      // tensor_dim0 = 6144 (lo16)
            g1[2] = 1 << 16;                 // tensor_dim0 hi = 0 ; tensor_dim1 = 1
            g1[3] = (int)(6144u << 16);      // tensor_dim1 hi = 0 ; tile_dim0 = 6144
            g1[4] = 1;                       // tile_dim1 = 1, tile_dim2 = 0
            g1[5] = 6144;                    // tensor_dim0_stride lo32
            g1[6] = 0;
            g1[7] = 0;
            i32x4 gz = {0, 0, 0, 0};
#if defined(__clang_major__) && __clang_major__ >= 23
            __builtin_amdgcn_tensor_load_to_lds(g0, g1, gz, gz, (i32x8)0, 0);
#else
            __builtin_amdgcn_tensor_load_to_lds(g0, g1, gz, gz, 0);
#endif
        }
#else
        // fallback: contiguous vectorized copy, 16 elements per thread
        const __bf16* src = Kg + (size_t)kk * KS + tid * 16;
        __bf16* dst = &KT[buf][0][0] + tid * 16;
        *(v8bf*)dst       = *(const v8bf*)src;
        *(v8bf*)(dst + 8) = *(const v8bf*)(src + 8);
#endif
    };

    stage(0, 0);
#if USE_TDM
    if (wid == 0) __builtin_amdgcn_s_wait_tensorcnt(0);
#endif
    __syncthreads();

    for (int it = 0; it < 16; ++it) {
        const int kk = it * 32;
        const int cur = it & 1;
        if (it + 1 < 16) stage(kk + 32, cur ^ 1);   // overlap next-chunk DMA

        v16bf af;   // A = Q rows t, straight from global bf16 (two b128 chunks)
        {
            const __bf16* ap = Qb + (size_t)(16 * wid + lrow) * KD + kk + 8 * khalf;
            __builtin_prefetch((const void*)(ap + 32), 0, 3);  // next chunk of Q
            v8bf lo = *(const v8bf*)ap;
            v8bf hi = *(const v8bf*)(ap + 16);
            #pragma unroll
            for (int i = 0; i < 8; ++i) { af[i] = lo[i]; af[8 + i] = hi[i]; }
        }
        #pragma unroll
        for (int j = 0; j < 12; ++j) {
            v16bf bfv;
            #pragma unroll
            for (int i = 0; i < 8; ++i) { bfv[i] = KT[cur][8 * khalf + i][16 * j + lrow]; bfv[8 + i] = KT[cur][16 + 8 * khalf + i][16 * j + lrow]; }
            acc[j] = __builtin_amdgcn_wmma_f32_16x16x32_bf16(
                false, af, false, bfv, (short)0, acc[j], false, false);
        }

#if USE_TDM
        if (wid == 0) __builtin_amdgcn_s_wait_tensorcnt(0);
#endif
        __syncthreads();
    }

    // mask: C[t][s] *= sigmoid(se[s][t]); t contiguous over r
    const int t0 = 16 * wid + 8 * khalf;
    #pragma unroll
    for (int j = 0; j < 12; ++j) {
        int s = 16 * j + lrow;
        const float* sp = sig + ((size_t)l * KS + s) * KS + t0;
        float4 sa = *(const float4*)sp;
        float4 sb = *(const float4*)(sp + 4);
        acc[j][0] *= sa.x; acc[j][1] *= sa.y; acc[j][2] *= sa.z; acc[j][3] *= sa.w;
        acc[j][4] *= sb.x; acc[j][5] *= sb.y; acc[j][6] *= sb.z; acc[j][7] *= sb.w;
    }

    // row-max (max over s) -> cat[0:192] indexed by t
    #pragma unroll
    for (int r = 0; r < 8; ++r) {
        float v = acc[0][r];
        #pragma unroll
        for (int j = 1; j < 12; ++j) v = fmaxf(v, acc[j][r]);
        v = fmaxf(v, __shfl_xor(v, 1, 32));
        v = fmaxf(v, __shfl_xor(v, 2, 32));
        v = fmaxf(v, __shfl_xor(v, 4, 32));
        v = fmaxf(v, __shfl_xor(v, 8, 32));
        if (lrow == 0) catv[t0 + r] = v;
    }

    // col-max (max over t) via order-preserving u32 keys
    if (tid < KS) cmkey[tid] = 0u;
    __syncthreads();
    #pragma unroll
    for (int j = 0; j < 12; ++j) {
        float v = acc[j][0];
        #pragma unroll
        for (int r = 1; r < 8; ++r) v = fmaxf(v, acc[j][r]);
        unsigned bits = __float_as_uint(v);
        unsigned key = (bits & 0x80000000u) ? ~bits : (bits | 0x80000000u);
        atomicMax(&cmkey[16 * j + lrow], key);
    }
    __syncthreads();
    if (tid < KS) {
        unsigned key = cmkey[tid];
        unsigned bits = (key & 0x80000000u) ? (key & 0x7FFFFFFFu) : ~key;
        catv[KS + tid] = __uint_as_float(bits);
    }
    __syncthreads();

    // write cat + BN1 partial sums over all 384 values
    float* catp = cat + ((size_t)l * 256 + qk) * (2 * KS);
    float v = catv[tid];
    catp[tid] = v;
    float s1 = v, s2 = v * v;
    s1 += __shfl_xor(s1, 1, 32);  s2 += __shfl_xor(s2, 1, 32);
    s1 += __shfl_xor(s1, 2, 32);  s2 += __shfl_xor(s2, 2, 32);
    s1 += __shfl_xor(s1, 4, 32);  s2 += __shfl_xor(s2, 4, 32);
    s1 += __shfl_xor(s1, 8, 32);  s2 += __shfl_xor(s2, 8, 32);
    s1 += __shfl_xor(s1, 16, 32); s2 += __shfl_xor(s2, 16, 32);
    if (lane == 0) { red[wid] = s1; red[12 + wid] = s2; }
    __syncthreads();
    if (tid == 0) {
        float t1 = 0.f, t2 = 0.f;
        #pragma unroll
        for (int w = 0; w < 12; ++w) { t1 += red[w]; t2 += red[12 + w]; }
        atomicAdd(&bn1[l * 2 + 0], t1);
        atomicAdd(&bn1[l * 2 + 1], t2);
    }
}

// --------------------------------- fc2: y = bn1(x) @ w2^T + b2, + BN2 partials
// grid: x = (512/64)*(2048/64)=256, y=l ; block=128 (4 waves)
__global__ __launch_bounds__(128)
void k_fc2(const float* __restrict__ cat, const float* __restrict__ bn1,
           const float* __restrict__ g1, const float* __restrict__ b1,
           const float* __restrict__ w2, const float* __restrict__ b2,
           float* __restrict__ y, float* __restrict__ c2sum, float* __restrict__ c2sq) {
    const int mb = blockIdx.x & 7, nb = blockIdx.x >> 3;
    const int l = blockIdx.y;
    const int tid = threadIdx.x;
    const int wid = tid >> 5, lane = tid & 31;
    const int khalf = lane >> 4, lrow = lane & 15;
    const int m0 = mb * 64, n0 = nb * 64;

    const float inv_n1 = 1.0f / 98304.0f;
    float mu  = bn1[l * 2 + 0] * inv_n1;
    float var = bn1[l * 2 + 1] * inv_n1 - mu * mu;
    float sc  = g1[l] * rsqrtf(var + KEPS);
    float sh  = b1[l] - mu * sc;

    const float* Abase = cat + (size_t)l * 98304;        // flat [512][192]
    const float* Bbase = w2 + (size_t)l * KDFF * KS;     // [n][k]

    __shared__ __attribute__((aligned(16))) __bf16 As[64][40];
    __shared__ __attribute__((aligned(16))) __bf16 Bs[32][72];
    v8f acc[4] = {};

    for (int kk = 0; kk < KS; kk += 32) {
        { // stage A (row-contiguous): BN1 fused, two b128 stores per thread
            int m = tid >> 1, kstart = (tid & 1) * 16;
            const float* src = Abase + (size_t)(m0 + m) * KS + kk + kstart;
            v8bf p0, p1;
            #pragma unroll
            for (int i = 0; i < 8; ++i) p0[i] = (__bf16)(src[i] * sc + sh);
            #pragma unroll
            for (int i = 0; i < 8; ++i) p1[i] = (__bf16)(src[8 + i] * sc + sh);
            *(v8bf*)(&As[m][kstart])     = p0;
            *(v8bf*)(&As[m][kstart + 8]) = p1;
        }
        { // stage B transposed: pair of n-rows x 8 k, packed b32 stores
            int np = tid >> 2;             // 0..31 -> n = 2*np, 2*np+1
            int kstart = (tid & 3) * 8;    // 0..24
            const float* s0 = Bbase + (size_t)(n0 + 2 * np) * KS + kk + kstart;
            const float* s1 = s0 + KS;
            #pragma unroll
            for (int i = 0; i < 8; ++i) {
                v2bf p; p[0] = (__bf16)s0[i]; p[1] = (__bf16)s1[i];
                *(v2bf*)(&Bs[kstart + i][2 * np]) = p;
            }
        }
        __syncthreads();
        v16bf af;
        {
            const __bf16* ap = &As[16 * wid + lrow][0];
            v8bf lo = *(const v8bf*)(ap + 8 * khalf);
            v8bf hi = *(const v8bf*)(ap + 16 + 8 * khalf);
            #pragma unroll
            for (int i = 0; i < 8; ++i) { af[i] = lo[i]; af[8 + i] = hi[i]; }
        }
        #pragma unroll
        for (int j = 0; j < 4; ++j) {
            v16bf bfv;
            #pragma unroll
            for (int i = 0; i < 8; ++i) { bfv[i] = Bs[8 * khalf + i][16 * j + lrow]; bfv[8 + i] = Bs[16 + 8 * khalf + i][16 * j + lrow]; }
            acc[j] = __builtin_amdgcn_wmma_f32_16x16x32_bf16(
                false, af, false, bfv, (short)0, acc[j], false, false);
        }
        __syncthreads();
    }

    #pragma unroll
    for (int j = 0; j < 4; ++j) {
        int n = n0 + 16 * j + lrow;
        float bias = b2[l * KDFF + n];
        float s1 = 0.f, s2 = 0.f;
        #pragma unroll
        for (int r = 0; r < 8; ++r) {
            int m = m0 + 16 * wid + 8 * khalf + r;
            float v = acc[j][r] + bias;
            y[((size_t)l * 512 + m) * KDFF + n] = v;
            s1 += v; s2 += v * v;
        }
        atomicAdd(&c2sum[l * KDFF + n], s1);
        atomicAdd(&c2sq[l * KDFF + n], s2);
    }
}

// --------------------- BN2 apply + relu + fc3 dot per row ; grid x=512, y=l
__global__ __launch_bounds__(256)
void k_fc3(const float* __restrict__ y, const float* __restrict__ c2sum,
           const float* __restrict__ c2sq, const float* __restrict__ g2,
           const float* __restrict__ bb2, const float* __restrict__ w3,
           const float* __restrict__ b3, float* __restrict__ fc3out) {
    const int m = blockIdx.x, l = blockIdx.y;
    const int tid = threadIdx.x;
    const int wid = tid >> 5, lane = tid & 31;
    const float* yr = y + ((size_t)l * 512 + m) * KDFF;
    const float inv_r = 1.0f / 512.0f;

    float accu = 0.f;
    for (int n = tid; n < KDFF; n += 256) {
        float mu  = c2sum[l * KDFF + n] * inv_r;
        float var = c2sq[l * KDFF + n] * inv_r - mu * mu;
        float v = g2[l * KDFF + n] * (yr[n] - mu) * rsqrtf(var + KEPS) + bb2[l * KDFF + n];
        v = fmaxf(v, 0.f);
        accu += v * w3[l * KDFF + n];
    }
    accu += __shfl_xor(accu, 1, 32);
    accu += __shfl_xor(accu, 2, 32);
    accu += __shfl_xor(accu, 4, 32);
    accu += __shfl_xor(accu, 8, 32);
    accu += __shfl_xor(accu, 16, 32);
    __shared__ float red[8];
    if (lane == 0) red[wid] = accu;
    __syncthreads();
    if (tid == 0) {
        float t = 0.f;
        #pragma unroll
        for (int w = 0; w < 8; ++w) t += red[w];
        fc3out[l * 512 + m] = t + b3[l];
    }
}

// ------------------ pair-sum + BN3 + layer accumulation + pair labels ; 1 block
__global__ __launch_bounds__(256)
void k_final(const float* __restrict__ fc3out, const float* __restrict__ g3,
             const float* __restrict__ bb3, const int* __restrict__ targets,
             float* __restrict__ out) {
    const int t = threadIdx.x;
    const int wid = t >> 5, lane = t & 31;
    __shared__ float red[16];
    __shared__ float stats[2];
    float score = 0.f;
    for (int l = 0; l < KL; ++l) {
        float z = fc3out[l * 512 + 2 * t] + fc3out[l * 512 + 2 * t + 1];
        float s1 = z, s2 = z * z;
        s1 += __shfl_xor(s1, 1, 32);  s2 += __shfl_xor(s2, 1, 32);
        s1 += __shfl_xor(s1, 2, 32);  s2 += __shfl_xor(s2, 2, 32);
        s1 += __shfl_xor(s1, 4, 32);  s2 += __shfl_xor(s2, 4, 32);
        s1 += __shfl_xor(s1, 8, 32);  s2 += __shfl_xor(s2, 8, 32);
        s1 += __shfl_xor(s1, 16, 32); s2 += __shfl_xor(s2, 16, 32);
        if (lane == 0) { red[wid] = s1; red[8 + wid] = s2; }
        __syncthreads();
        if (t == 0) {
            float a = 0.f, b = 0.f;
            #pragma unroll
            for (int w = 0; w < 8; ++w) { a += red[w]; b += red[8 + w]; }
            stats[0] = a * (1.0f / 256.0f);
            stats[1] = b * (1.0f / 256.0f);
        }
        __syncthreads();
        float mu = stats[0];
        float var = stats[1] - mu * mu;
        score += g3[l] * (z - mu) * rsqrtf(var + KEPS) + bb3[l];
        __syncthreads();
    }
    out[t] = score;
    out[256 + t] = (targets[t >> 4] == targets[t & 15]) ? 1.0f : 0.0f;
}

extern "C" void kernel_launch(void* const* d_in, const int* in_sizes, int n_in,
                              void* d_out, int out_size, void* d_ws, size_t ws_size,
                              hipStream_t stream) {
    (void)in_sizes; (void)n_in; (void)out_size; (void)ws_size;
    const float* q_feat      = (const float*)d_in[0];
    const float* g_feat      = (const float*)d_in[1];
    const int*   targets     = (const int*)d_in[2];
    const float* score_embed = (const float*)d_in[3];
    const float* fc0_w       = (const float*)d_in[4];
    const float* fc0_b       = (const float*)d_in[5];
    const float* bn1_g       = (const float*)d_in[6];
    const float* bn1_b       = (const float*)d_in[7];
    const float* fc2_w       = (const float*)d_in[8];
    const float* fc2_b       = (const float*)d_in[9];
    const float* bn2_g       = (const float*)d_in[10];
    const float* bn2_b       = (const float*)d_in[11];
    const float* fc3_w       = (const float*)d_in[12];
    const float* fc3_b       = (const float*)d_in[13];
    const float* bn3_g       = (const float*)d_in[14];
    const float* bn3_b       = (const float*)d_in[15];
    float* out = (float*)d_out;

    char* ws = (char*)d_ws;
    size_t off = 0;
    auto carve = [&](size_t bytes) {
        void* p = ws + off;
        off = (off + bytes + 255) & ~(size_t)255;
        return p;
    };
    float*  sig   = (float*)carve((size_t)KL * KS * KS * 4);          // 1.18 MB
    __bf16* Qp    = (__bf16*)carve((size_t)KL * KBS * KS * KD * 2);   // 25.2 MB
    __bf16* KpT   = (__bf16*)carve((size_t)KL * KBS * KS * KD * 2);   // 25.2 MB
    float*  cat   = (float*)carve((size_t)KL * 256 * 2 * KS * 4);     // 3.1 MB
    float*  ybuf  = (float*)carve((size_t)KL * 512 * KDFF * 4);       // 33.6 MB
    float*  bn1s  = (float*)carve((size_t)KL * 2 * 4);
    float*  c2sum = (float*)carve((size_t)KL * KDFF * 4);
    float*  c2sq  = (float*)carve((size_t)KL * KDFF * 4);
    float*  fc3o  = (float*)carve((size_t)KL * 512 * 4);

    hipMemsetAsync(bn1s,  0, (size_t)KL * 2 * 4, stream);
    hipMemsetAsync(c2sum, 0, (size_t)KL * KDFF * 4, stream);
    hipMemsetAsync(c2sq,  0, (size_t)KL * KDFF * 4, stream);

    int nsig = KL * KS * KS;
    k_sigmoid<<<(nsig + 255) / 256, 256, 0, stream>>>(score_embed, sig, nsig);
    k_project<<<dim3(24, KBS, 2 * KL), 128, 0, stream>>>(q_feat, g_feat, fc0_w, fc0_b, Qp, KpT);
    k_corr<<<dim3(256, KL), 384, 0, stream>>>(Qp, KpT, sig, cat, bn1s);
    k_fc2<<<dim3(256, KL), 128, 0, stream>>>(cat, bn1s, bn1_g, bn1_b, fc2_w, fc2_b,
                                             ybuf, c2sum, c2sq);
    k_fc3<<<dim3(512, KL), 256, 0, stream>>>(ybuf, c2sum, c2sq, bn2_g, bn2_b,
                                             fc3_w, fc3_b, fc3o);
    k_final<<<1, 256, 0, stream>>>(fc3o, bn3_g, bn3_b, targets, out);
}
